// GAT_51711406243963
// MI455X (gfx1250) — compile-verified, matrix operand order checked by reading, not run on hardware
//
#include <hip/hip_runtime.h>
#include <hip/hip_bf16.h>
#include <math.h>

// CDNA5 / gfx1250: wave32, WMMA 16x16x32 f16->f32, TDM async tensor loads
typedef __attribute__((ext_vector_type(16))) _Float16 v16h;
typedef __attribute__((ext_vector_type(8)))  _Float16 v8h;
typedef __attribute__((ext_vector_type(8)))  float    v8f;
typedef __attribute__((ext_vector_type(4)))  unsigned u32x4;
typedef __attribute__((ext_vector_type(8)))  int      i32x8;
typedef __attribute__((ext_vector_type(4)))  int      i32x4;

__device__ __forceinline__ float sigm_(float x) { return 1.f / (1.f + __expf(-x)); }
__device__ __forceinline__ float tanhfast_(float x) {
    float e = __expf(2.f * x);
    return (e - 1.f) / (e + 1.f);
}
__device__ __forceinline__ float leaky_(float x, float s) { return x >= 0.f ? x : s * x; }
__device__ __forceinline__ unsigned f2ord_(float f) {
    unsigned u = __float_as_uint(f);
    return (u & 0x80000000u) ? ~u : (u | 0x80000000u);
}
__device__ __forceinline__ float ord2f_(unsigned o) {
    unsigned u = (o & 0x80000000u) ? (o ^ 0x80000000u) : ~o;
    return __uint_as_float(u);
}
__device__ __forceinline__ float wred_(float v) {
    for (int o = 16; o; o >>= 1) v += __shfl_down(v, o, 32);
    return v;
}

// Assemble a WMMA A-fragment (ISA 7.12.2 16-bit A 16x32) from a row-major
// LDS tile: per lane it is two contiguous 8-half (16B) runs -> 2x ds_load_b128.
__device__ __forceinline__ v16h afrag_ld_(const _Float16* rowp /*&T[am][kt*32+kb]*/) {
    v8h lo = *(const v8h*)rowp;         // K = kb .. kb+7
    v8h hi = *(const v8h*)(rowp + 16);  // K = 16+kb .. 16+kb+7
    v16h a;
#pragma unroll
    for (int i = 0; i < 8; ++i) { a[i] = lo[i]; a[8 + i] = hi[i]; }
    return a;
}

// ---------------------------------------------------------------------------
// P = emb_url @ Wih + b (per direction): only 128 distinct URLs, so the LSTM
// input projection collapses into a 128x512 lookup table.
// ---------------------------------------------------------------------------
__global__ void gnn_compute_P(const float* __restrict__ emb,
                              const float* __restrict__ WihF, const float* __restrict__ bF,
                              const float* __restrict__ WihB, const float* __restrict__ bB,
                              float* __restrict__ Pf, float* __restrict__ Pb) {
    int idx = blockIdx.x * blockDim.x + threadIdx.x;
    if (idx >= 2 * 128 * 512) return;
    int dir = idx / (128 * 512);
    int rem = idx - dir * 128 * 512;
    int u = rem >> 9, g = rem & 511;
    const float* W = dir ? WihB : WihF;
    const float* b = dir ? bB : bF;
    float s = b[g];
    for (int k = 0; k < 128; ++k) s += emb[u * 128 + k] * W[k * 512 + g];
    (dir ? Pb : Pf)[rem] = s;
}

// ---------------------------------------------------------------------------
// Pre-swizzle a row-major KxN f32 weight into WMMA B-fragment order (f16):
// [ntile][ktile][lane][16 halfs]; a wave's fragment load is 32B/lane (2xb128).
// ---------------------------------------------------------------------------
__global__ void gnn_swizzle_B(const float* __restrict__ src, _Float16* __restrict__ dst,
                              int K, int N) {
    int KT = K >> 5, NT = N >> 4;
    int total = KT * NT * 32;
    int idx = blockIdx.x * blockDim.x + threadIdx.x;
    if (idx >= total) return;
    int lane = idx & 31;
    int f = idx >> 5;
    int kt = f % KT, nt = f / KT;
    int n = nt * 16 + (lane & 15);
    int kb = (lane >> 4) * 8;
    _Float16 tmp[16];
#pragma unroll
    for (int i = 0; i < 8; ++i) {
        int k0 = kt * 32 + ((i < 4) ? (kb + 2 * i) : (16 + kb + 2 * (i - 4)));
        tmp[2 * i]     = (_Float16)src[(size_t)k0 * N + n];
        tmp[2 * i + 1] = (_Float16)src[(size_t)(k0 + 1) * N + n];
    }
#pragma unroll
    for (int j = 0; j < 16; ++j) dst[(size_t)idx * 16 + j] = tmp[j];
}

// ---------------------------------------------------------------------------
// BiLSTM recurrence: WG (4 waves) owns a 16-node tile; h/c/gates in LDS;
// gates(16x512) = P[token] (C-init gather) + h(16x128 f16) @ Whh (L2-hot
// swizzled f16 fragments). 128 v_wmma per step per WG.
// ---------------------------------------------------------------------------
__global__ __launch_bounds__(128)
void gnn_lstm_wmma(const int* __restrict__ toks, const unsigned char* __restrict__ mask,
                   const float* __restrict__ Pf, const float* __restrict__ Pb,
                   const _Float16* __restrict__ WswF, const _Float16* __restrict__ WswB,
                   float* __restrict__ hout, int N, int seq) {
    __shared__ __align__(16) _Float16 Hh[16][128];   // f16 h (WMMA A source)
    __shared__ float    Hf[16][128];                 // f32 h (output precision)
    __shared__ float    Cc[16][128];
    __shared__ float    Gt[16][512];                 // gates
    __shared__ int      Tok[16][32];
    __shared__ unsigned char Msk[16][32];

    int dir = blockIdx.y;
    const float*    P   = dir ? Pb : Pf;
    const _Float16* Wsw = dir ? WswB : WswF;
    int node0 = blockIdx.x * 16;
    int tid = threadIdx.x, lane = tid & 31, wv = tid >> 5;

    for (int idx = tid; idx < 16 * 32; idx += 128) {
        int r = idx >> 5, t = idx & 31;
        int ts = dir ? (seq - 1 - t) : t;
        int node = node0 + r;
        Tok[r][t] = (node < N) ? toks[(size_t)node * seq + ts] : 0;
        Msk[r][t] = (node < N) ? mask[(size_t)node * seq + ts] : 0;
    }
    for (int idx = tid; idx < 16 * 128; idx += 128) {
        int r = idx >> 7, j = idx & 127;
        Hh[r][j] = (_Float16)0.f; Hf[r][j] = 0.f; Cc[r][j] = 0.f;
    }

    const v16h* Bf = (const v16h*)Wsw;
    int am = lane & 15, kb = (lane >> 4) * 8;
    int nl = lane & 15;
    int mhi = 8 * (lane >> 4);

    for (int t = 0; t < seq; ++t) {
        __syncthreads();
        // A fragments (shared by all 8 n-tiles this wave computes): 2x b128 each
        v16h afrag[4];
#pragma unroll
        for (int kt = 0; kt < 4; ++kt)
            afrag[kt] = afrag_ld_(&Hh[am][kt * 32 + kb]);

        for (int s = 0; s < 8; ++s) {
            int nt = wv * 8 + s;
            v8f acc;
#pragma unroll
            for (int i = 0; i < 8; ++i)            // C init = P[token_m][col]
                acc[i] = P[(size_t)Tok[i + mhi][t] * 512 + nt * 16 + nl];
#pragma unroll
            for (int kt = 0; kt < 4; ++kt) {
                v16h b = Bf[(size_t)(nt * 4 + kt) * 32 + lane];
                if (kt < 3)
                    __builtin_prefetch(&Bf[(size_t)(nt * 4 + kt + 1) * 32 + lane], 0, 3);
                acc = __builtin_amdgcn_wmma_f32_16x16x32_f16(
                    false, afrag[kt], false, b, (short)0, acc, false, false);
            }
#pragma unroll
            for (int i = 0; i < 8; ++i) Gt[i + mhi][nt * 16 + nl] = acc[i];
        }
        __syncthreads();
        for (int e = tid; e < 16 * 128; e += 128) {
            int r = e >> 7, j = e & 127;
            float gi = Gt[r][j], gf = Gt[r][128 + j], gg = Gt[r][256 + j], go = Gt[r][384 + j];
            float c2 = sigm_(gf) * Cc[r][j] + sigm_(gi) * tanhfast_(gg);
            float h2 = sigm_(go) * tanhfast_(c2);
            if (Msk[r][t]) { Cc[r][j] = c2; Hf[r][j] = h2; Hh[r][j] = (_Float16)h2; }
        }
    }
    __syncthreads();
    for (int e = tid; e < 16 * 128; e += 128) {
        int r = e >> 7, j = e & 127;
        int node = node0 + r;
        if (node < N) hout[(size_t)node * 256 + dir * 128 + j] = Hf[r][j];
    }
}

// ---------------------------------------------------------------------------
// Generic WMMA GEMM. The A tile (16 x K, f32, row stride lda) is staged from
// global to LDS by the Tensor Data Mover (2D D#: tile K x 16, dim0_stride=lda,
// OOB rows read zero), then converted to f16. B comes pre-swizzled from L2.
// 4 waves: 16(M) x 64(N) tile. K%32==0, N%64==0 for all uses.
// ---------------------------------------------------------------------------
__global__ __launch_bounds__(128)
void gnn_gemm_wmma(const float* __restrict__ A, int lda, int M, int K,
                   const _Float16* __restrict__ Bsw, int N,
                   const float* __restrict__ bias, float slope,
                   float* __restrict__ C, int ldc, int colOff) {
    __shared__ __align__(16) float    Af32[16][544];
    __shared__ __align__(16) _Float16 As[16][544];
    int row0 = blockIdx.x * 16;
    int tid = threadIdx.x, lane = tid & 31, wv = tid >> 5;

    // ---- TDM: async 2D tile load global -> LDS (wave 0 issues; EXEC ignored)
    if (wv == 0) {
        unsigned lds0 = (unsigned)(uintptr_t)&Af32[0][0];
        unsigned long long ga =
            (unsigned long long)(uintptr_t)(A + (size_t)row0 * lda);
        unsigned rows = (unsigned)(M - row0);          // OOB rows -> zero
        u32x4 g0;
        g0[0] = 1u;                                    // count=1, user mode
        g0[1] = lds0;                                  // lds_addr
        g0[2] = (unsigned)ga;                          // global_addr[31:0]
        g0[3] = ((unsigned)(ga >> 32) & 0x01FFFFFFu)   // global_addr[56:32]
              | 0x80000000u;                           // type=2 ("image")
        i32x8 g1;
        g1[0] = (int)(2u << 16);                       // data_size = 4B
        g1[1] = (int)(((unsigned)K & 0xFFFFu) << 16);  // tensor_dim0[15:0]
        g1[2] = (int)(((unsigned)K >> 16) | ((rows & 0xFFFFu) << 16)); // dim0 hi | dim1 lo
        g1[3] = (int)(((rows >> 16) & 0xFFFFu) | (((unsigned)K & 0xFFFFu) << 16)); // dim1 hi | tile_dim0=K
        g1[4] = 16;                                    // tile_dim1 = 16 rows
        g1[5] = lda;                                   // tensor_dim0_stride
        g1[6] = 0; g1[7] = 0;
        i32x4 z4 = {0, 0, 0, 0};
#if defined(__clang_major__) && (__clang_major__ >= 23)
        i32x8 z8 = {0, 0, 0, 0, 0, 0, 0, 0};
        __builtin_amdgcn_tensor_load_to_lds(g0, g1, z4, z4, z8, 0);
#else
        __builtin_amdgcn_tensor_load_to_lds(g0, g1, z4, z4, 0);
#endif
        __builtin_amdgcn_s_wait_tensorcnt(0);
    }
    __syncthreads();
    for (int idx = tid; idx < 16 * K; idx += 128) {
        int r = idx / K, k = idx - r * K;
        As[r][k] = (_Float16)Af32[r][k];
    }
    __syncthreads();

    int nt = blockIdx.y * 4 + wv;
    int KT = K >> 5;
    const v16h* Bf = (const v16h*)Bsw;
    int am = lane & 15, kb = (lane >> 4) * 8;
    int nl = lane & 15, mhi = 8 * (lane >> 4);
    int col = nt * 16 + nl;

    v8f acc;
#pragma unroll
    for (int i = 0; i < 8; ++i) acc[i] = bias ? bias[col] : 0.f;

    for (int kt = 0; kt < KT; ++kt) {
        v16h a = afrag_ld_(&As[am][kt * 32 + kb]);
        v16h b = Bf[(size_t)(nt * KT + kt) * 32 + lane];
        if (kt + 1 < KT)
            __builtin_prefetch(&Bf[(size_t)(nt * KT + kt + 1) * 32 + lane], 0, 3);
        acc = __builtin_amdgcn_wmma_f32_16x16x32_f16(
            false, a, false, b, (short)0, acc, false, false);
    }
#pragma unroll
    for (int i = 0; i < 8; ++i) {
        int row = row0 + i + mhi;
        if (row < M) C[(size_t)row * ldc + colOff + col] = leaky_(acc[i], slope);
    }
}

// ---------------------------------------------------------------------------
__global__ void gnn_concat_fill(const int* __restrict__ c, const int* __restrict__ co,
                                const int* __restrict__ sl, const float* __restrict__ ip,
                                const float* __restrict__ Ecat, const float* __restrict__ Ecty,
                                const float* __restrict__ Esec,
                                float* __restrict__ hcat, int N) {
    int idx = blockIdx.x * blockDim.x + threadIdx.x;
    if (idx >= N * 416) return;
    int node = idx / 416, j = idx - node * 416;
    float v;
    if (j < 128)      v = Ecat[c[node] * 128 + j];
    else if (j < 256) v = Ecty[co[node] * 128 + (j - 128)];
    else if (j < 384) v = Esec[sl[node] * 128 + (j - 256)];
    else              v = ip[(size_t)node * 32 + (j - 384)];
    hcat[(size_t)node * 544 + 128 + j] = v;
}

// ---------------------------------------------------------------------------
// GAT edge-softmax machinery (heads = 1)
// ---------------------------------------------------------------------------
__global__ void gnn_eler(const float* __restrict__ ft, const float* __restrict__ al,
                         const float* __restrict__ ar, float* __restrict__ el,
                         float* __restrict__ er, int N, int D) {
    int w = (blockIdx.x * blockDim.x + threadIdx.x) >> 5;
    int lane = threadIdx.x & 31;
    if (w >= N) return;
    float sl = 0.f, sr = 0.f;
    for (int j = lane; j < D; j += 32) {
        float f = ft[(size_t)w * D + j];
        sl += f * al[j]; sr += f * ar[j];
    }
    sl = wred_(sl); sr = wred_(sr);
    if (lane == 0) { el[w] = sl; er[w] = sr; }
}

__global__ void gnn_node_init(unsigned* __restrict__ emaxo, float* __restrict__ den, int N) {
    int i = blockIdx.x * blockDim.x + threadIdx.x;
    if (i < N) { emaxo[i] = 0u; den[i] = 0.f; }
}

__global__ void gnn_edge_pass1(const int* __restrict__ src, const int* __restrict__ dst,
                               const float* __restrict__ el, const float* __restrict__ er,
                               float* __restrict__ earr, unsigned* __restrict__ emaxo, int E) {
    int e = blockIdx.x * blockDim.x + threadIdx.x;
    if (e >= E) return;
    float v = leaky_(el[src[e]] + er[dst[e]], 0.2f);
    earr[e] = v;
    atomicMax(&emaxo[dst[e]], f2ord_(v));
}

__global__ void gnn_node_decode(const unsigned* __restrict__ emaxo, float* __restrict__ emax, int N) {
    int i = blockIdx.x * blockDim.x + threadIdx.x;
    if (i >= N) return;
    float v = ord2f_(emaxo[i]);
    emax[i] = (v == v && fabsf(v) != INFINITY) ? v : 0.f;  // empty segment -> 0
}

__global__ void gnn_edge_pass2(const int* __restrict__ dst, const float* __restrict__ emax,
                               float* __restrict__ earr, float* __restrict__ den, int E) {
    int e = blockIdx.x * blockDim.x + threadIdx.x;
    if (e >= E) return;
    int d = dst[e];
    float ex = __expf(earr[e] - emax[d]);
    earr[e] = ex;
    atomicAdd(&den[d], ex);
}

__global__ void gnn_rst_init(float* __restrict__ rst, const float* __restrict__ bias, int N, int D) {
    int idx = blockIdx.x * blockDim.x + threadIdx.x;
    if (idx >= N * D) return;
    rst[idx] = bias[idx % D];
}

__global__ void gnn_edge_pass3(const int* __restrict__ src, const int* __restrict__ dst,
                               const float* __restrict__ earr, const float* __restrict__ den,
                               const float* __restrict__ ft, float* __restrict__ rst,
                               int E, int D) {
    int w = (blockIdx.x * blockDim.x + threadIdx.x) >> 5;
    int lane = threadIdx.x & 31;
    if (w >= E) return;
    int s = src[w], d = dst[w];
    float alpha = earr[w] / den[d];
    const float* fs = ft + (size_t)s * D;
    float* rd = rst + (size_t)d * D;
    for (int j = lane; j < D; j += 32) atomicAdd(&rd[j], fs[j] * alpha);
}

__global__ void gnn_act_inplace(float* __restrict__ x, float slope, long long n) {
    long long i = (long long)blockIdx.x * blockDim.x + threadIdx.x;
    if (i < n) x[i] = leaky_(x[i], slope);
}

__global__ void gnn_score(const int* __restrict__ ss, const int* __restrict__ sd,
                          const float* __restrict__ h, float* __restrict__ out, int S) {
    int w = (blockIdx.x * blockDim.x + threadIdx.x) >> 5;
    int lane = threadIdx.x & 31;
    if (w >= S) return;
    const float* a = h + (size_t)ss[w] * 128;
    const float* b = h + (size_t)sd[w] * 128;
    float s = 0.f;
    for (int j = lane; j < 128; j += 32) s += a[j] * b[j];
    s = wred_(s);
    if (lane == 0) out[w] = s;
}

// ---------------------------------------------------------------------------
static inline int cdiv_(int a, int b) { return (a + b - 1) / b; }

extern "C" void kernel_launch(void* const* d_in, const int* in_sizes, int n_in,
                              void* d_out, int out_size, void* d_ws, size_t ws_size,
                              hipStream_t stream) {
    const int*           toks = (const int*)d_in[0];
    const unsigned char* msk  = (const unsigned char*)d_in[1];
    const int*   in_c  = (const int*)d_in[2];
    const int*   in_co = (const int*)d_in[3];
    const int*   in_sl = (const int*)d_in[4];
    const float* in_ip = (const float*)d_in[5];
    const int*   esrc  = (const int*)d_in[6];
    const int*   edst  = (const int*)d_in[7];
    const int*   ssrc  = (const int*)d_in[8];
    const int*   sdst  = (const int*)d_in[9];
    const float* emb_url = (const float*)d_in[10];
    const float* emb_cat = (const float*)d_in[11];
    const float* emb_cty = (const float*)d_in[12];
    const float* emb_sec = (const float*)d_in[13];
    const float* Wih_f = (const float*)d_in[14];
    const float* Whh_f = (const float*)d_in[15];
    const float* b_f   = (const float*)d_in[16];
    const float* Wih_b = (const float*)d_in[17];
    const float* Whh_b = (const float*)d_in[18];
    const float* b_b   = (const float*)d_in[19];
    const float* fc_W  = (const float*)d_in[20];
    const float* fc_b  = (const float*)d_in[21];
    const float* g0W  = (const float*)d_in[22];
    const float* g0al = (const float*)d_in[23];
    const float* g0ar = (const float*)d_in[24];
    const float* g0b  = (const float*)d_in[25];
    const float* g1W  = (const float*)d_in[26];
    const float* g1al = (const float*)d_in[27];
    const float* g1ar = (const float*)d_in[28];
    const float* g1b  = (const float*)d_in[29];

    const int N   = in_sizes[2];
    const int SEQ = in_sizes[0] / N;
    const int E   = in_sizes[6];
    const int S   = in_sizes[8];

    // ---- workspace layout ----
    char* base = (char*)d_ws;
    size_t off = 0;
    auto alloc = [&](size_t bytes) -> char* {
        off = (off + 255) & ~(size_t)255;
        char* p = base + off;
        off += bytes;
        return p;
    };
    float*     Pf     = (float*)alloc(128 * 512 * 4);
    float*     Pb     = (float*)alloc(128 * 512 * 4);
    _Float16*  WswF   = (_Float16*)alloc(128 * 512 * 2);
    _Float16*  WswB   = (_Float16*)alloc(128 * 512 * 2);
    _Float16*  fcWsw  = (_Float16*)alloc(256 * 128 * 2);
    _Float16*  W0sw   = (_Float16*)alloc(544 * 256 * 2);
    _Float16*  W1sw   = (_Float16*)alloc(256 * 128 * 2);
    float*     hfb    = (float*)alloc((size_t)N * 256 * 4);   // [hf | hb]
    float*     hcat   = (float*)alloc((size_t)N * 544 * 4);
    float*     ftb    = (float*)alloc((size_t)N * 256 * 4);   // ft (both layers)
    float*     h1     = (float*)alloc((size_t)N * 256 * 4);
    float*     h2     = (float*)alloc((size_t)N * 128 * 4);
    float*     elb    = (float*)alloc((size_t)N * 4);
    float*     erb    = (float*)alloc((size_t)N * 4);
    unsigned*  emaxo  = (unsigned*)alloc((size_t)N * 4);
    float*     emaxf  = (float*)alloc((size_t)N * 4);
    float*     den    = (float*)alloc((size_t)N * 4);
    float*     earr   = (float*)alloc((size_t)E * 4);
    (void)ws_size;

    // ---- 1. P tables + swizzled f16 weights ----
    gnn_compute_P<<<cdiv_(2 * 128 * 512, 256), 256, 0, stream>>>(
        emb_url, Wih_f, b_f, Wih_b, b_b, Pf, Pb);
    gnn_swizzle_B<<<cdiv_((128 / 32) * (512 / 16) * 32, 256), 256, 0, stream>>>(Whh_f, WswF, 128, 512);
    gnn_swizzle_B<<<cdiv_((128 / 32) * (512 / 16) * 32, 256), 256, 0, stream>>>(Whh_b, WswB, 128, 512);
    gnn_swizzle_B<<<cdiv_((256 / 32) * (128 / 16) * 32, 256), 256, 0, stream>>>(fc_W, fcWsw, 256, 128);
    gnn_swizzle_B<<<cdiv_((544 / 32) * (256 / 16) * 32, 256), 256, 0, stream>>>(g0W, W0sw, 544, 256);
    gnn_swizzle_B<<<cdiv_((256 / 32) * (128 / 16) * 32, 256), 256, 0, stream>>>(g1W, W1sw, 256, 128);

    // ---- 2. BiLSTM (WMMA recurrence) ----
    {
        dim3 grid(cdiv_(N, 16), 2);
        gnn_lstm_wmma<<<grid, 128, 0, stream>>>(toks, msk, Pf, Pb, WswF, WswB, hfb, N, SEQ);
    }

    // ---- 3. FC merge into hcat[:,0:128] + concat fill ----
    {
        dim3 grid(cdiv_(N, 16), 128 / 64);
        gnn_gemm_wmma<<<grid, 128, 0, stream>>>(hfb, 256, N, 256, fcWsw, 128,
                                                fc_b, 0.01f, hcat, 544, 0);
    }
    gnn_concat_fill<<<cdiv_(N * 416, 256), 256, 0, stream>>>(
        in_c, in_co, in_sl, in_ip, emb_cat, emb_cty, emb_sec, hcat, N);

    // ---- 4. GAT layer 0: 544 -> 256, leaky 0.01 ----
    {
        dim3 grid(cdiv_(N, 16), 256 / 64);
        gnn_gemm_wmma<<<grid, 128, 0, stream>>>(hcat, 544, N, 544, W0sw, 256,
                                                nullptr, 1.0f, ftb, 256, 0);
    }
    gnn_eler<<<cdiv_(N * 32, 256), 256, 0, stream>>>(ftb, g0al, g0ar, elb, erb, N, 256);
    gnn_node_init<<<cdiv_(N, 256), 256, 0, stream>>>(emaxo, den, N);
    gnn_edge_pass1<<<cdiv_(E, 256), 256, 0, stream>>>(esrc, edst, elb, erb, earr, emaxo, E);
    gnn_node_decode<<<cdiv_(N, 256), 256, 0, stream>>>(emaxo, emaxf, N);
    gnn_edge_pass2<<<cdiv_(E, 256), 256, 0, stream>>>(edst, emaxf, earr, den, E);
    gnn_rst_init<<<cdiv_(N * 256, 256), 256, 0, stream>>>(h1, g0b, N, 256);
    gnn_edge_pass3<<<cdiv_(E * 32, 256), 256, 0, stream>>>(esrc, edst, earr, den, ftb, h1, E, 256);
    gnn_act_inplace<<<cdiv_(N * 256, 256), 256, 0, stream>>>(h1, 0.01f, (long long)N * 256);

    // ---- 5. GAT layer 1: 256 -> 128, no activation ----
    {
        dim3 grid(cdiv_(N, 16), 128 / 64);
        gnn_gemm_wmma<<<grid, 128, 0, stream>>>(h1, 256, N, 256, W1sw, 128,
                                                nullptr, 1.0f, ftb, 128, 0);
    }
    gnn_eler<<<cdiv_(N * 32, 256), 256, 0, stream>>>(ftb, g1al, g1ar, elb, erb, N, 128);
    gnn_node_init<<<cdiv_(N, 256), 256, 0, stream>>>(emaxo, den, N);
    gnn_edge_pass1<<<cdiv_(E, 256), 256, 0, stream>>>(esrc, edst, elb, erb, earr, emaxo, E);
    gnn_node_decode<<<cdiv_(N, 256), 256, 0, stream>>>(emaxo, emaxf, N);
    gnn_edge_pass2<<<cdiv_(E, 256), 256, 0, stream>>>(edst, emaxf, earr, den, E);
    gnn_rst_init<<<cdiv_(N * 128, 256), 256, 0, stream>>>(h2, g1b, N, 128);
    gnn_edge_pass3<<<cdiv_(E * 32, 256), 256, 0, stream>>>(esrc, edst, earr, den, ftb, h2, E, 128);

    // ---- 6. scores ----
    gnn_score<<<cdiv_(S * 32, 256), 256, 0, stream>>>(ssrc, sdst, h2, (float*)d_out, S);
}
